// OptimizedAttention_84679575208514
// MI455X (gfx1250) — 16-bit and 8-bit matrix operand order repaired and checked by reading, compiled, not run on hardware
//
#include <hip/hip_runtime.h>
#include <stdint.h>

#define BB 2
#define NN 2048
#define CC 1024
#define HH 16
#define DD 64
#define C3 (3 * CC)

typedef __attribute__((ext_vector_type(8)))  float    v8f;
typedef __attribute__((ext_vector_type(8)))  int      v8i;
typedef __attribute__((ext_vector_type(16))) _Float16 v16h;
typedef __attribute__((ext_vector_type(16))) __bf16   v16bf;

union U16h { uint4 u[2]; v16h v; };
union U16b { uint4 u[2]; v16bf v; };

__device__ __forceinline__ unsigned char f32_to_e4m3(float x) {
  unsigned int u = __float_as_uint(x);
  unsigned int s = (u >> 24) & 0x80u;
  float ax = fminf(fabsf(x), 448.0f);
  unsigned char r;
  if (ax < 0.015625f) {
    int m = (int)rintf(ax * 512.0f);
    r = (unsigned char)m;
  } else {
    unsigned int bits = __float_as_uint(ax);
    int e = (int)((bits >> 23) & 0xffu) - 127;
    unsigned int mant = bits & 0x7fffffu;
    unsigned int rm   = mant >> 20;
    unsigned int rest = mant & 0xfffffu;
    unsigned int up = (rest > 0x80000u) || (rest == 0x80000u && (rm & 1u));
    rm += up;
    if (rm >= 8u) { rm = 0; e += 1; }
    if (e > 8) { e = 8; rm = 6; }
    r = (unsigned char)(((e + 7) << 3) | rm);
  }
  return (unsigned char)(r | s);
}

__global__ __launch_bounds__(256) void cvt_f32_bf16(const float* __restrict__ src,
                                                    __bf16* __restrict__ dst, int n) {
  int i = blockIdx.x * 256 + threadIdx.x;
  if (i < n) dst[i] = (__bf16)src[i];
}

__global__ __launch_bounds__(256) void cvt_transpose_bf16(const float* __restrict__ src,
                                                          __bf16* __restrict__ dst,
                                                          int K, int Ncols) {
  int i = blockIdx.x * 256 + threadIdx.x;
  if (i >= K * Ncols) return;
  int k = i / Ncols, n2 = i - k * Ncols;
  dst[(size_t)n2 * K + k] = (__bf16)src[i];
}

__global__ __launch_bounds__(256) void gemm_bf16_wmma(const __bf16* __restrict__ A,
                                                      const __bf16* __restrict__ Bt,
                                                      float* __restrict__ Cout,
                                                      const float* __restrict__ bias,
                                                      int M, int Nc, int K) {
  int lane = threadIdx.x & 31;
  int wave = threadIdx.x >> 5;
  int tiles_n = Nc >> 6;
  int tile = blockIdx.x * 8 + wave;
  if (tile >= (M >> 6) * tiles_n) return;
  int tm = tile / tiles_n, tn = tile - tm * tiles_n;
  int l15 = lane & 15, h = lane >> 4;

  const __bf16* Abase = A  + (size_t)(tm * 64 + l15) * K;
  const __bf16* Bbase = Bt + (size_t)(tn * 64 + l15) * K;

  v8f acc[4][4];
  v8f zero = {};
#pragma unroll
  for (int mi = 0; mi < 4; ++mi)
#pragma unroll
    for (int ni = 0; ni < 4; ++ni) acc[mi][ni] = zero;

  for (int kc = 0; kc < K; kc += 32) {
    U16b a[4], b[4];
#pragma unroll
    for (int mi = 0; mi < 4; ++mi) {
      const __bf16* Ar = Abase + (size_t)mi * 16 * K + kc;
      a[mi].u[0] = *(const uint4*)(Ar + h * 8);
      a[mi].u[1] = *(const uint4*)(Ar + 16 + h * 8);
    }
#pragma unroll
    for (int ni = 0; ni < 4; ++ni) {
      const __bf16* Br = Bbase + (size_t)ni * 16 * K + kc;
      b[ni].u[0] = *(const uint4*)(Br + h * 8);
      b[ni].u[1] = *(const uint4*)(Br + 16 + h * 8);
    }
#pragma unroll
    for (int mi = 0; mi < 4; ++mi)
#pragma unroll
      for (int ni = 0; ni < 4; ++ni)
        acc[mi][ni] = __builtin_amdgcn_wmma_f32_16x16x32_bf16(
            false, a[mi].v, false, b[ni].v, (short)0, acc[mi][ni], false, false);
  }

#pragma unroll
  for (int ni = 0; ni < 4; ++ni) {
    int col = tn * 64 + ni * 16 + l15;
    float bv = bias ? bias[col] : 0.0f;
#pragma unroll
    for (int mi = 0; mi < 4; ++mi) {
#pragma unroll
      for (int r = 0; r < 8; ++r) {
        int row = tm * 64 + mi * 16 + r + 8 * h;
        Cout[(size_t)row * Nc + col] = acc[mi][ni][r] + bv;
      }
    }
  }
}

__global__ __launch_bounds__(256) void amax_scale_kernel(const float* __restrict__ qkv,
                                                         float* __restrict__ scales) {
  int which = blockIdx.x;
  int bh = which >> 1, t = which & 1;
  int b = bh / HH, hh = bh - b * HH;
  __shared__ float red[256];
  const float* base = qkv + (size_t)b * NN * C3 + t * CC + hh * DD;
  float m = 0.0f;
  for (int i = threadIdx.x; i < NN * DD; i += 256) {
    int n = i >> 6, d = i & 63;
    m = fmaxf(m, fabsf(base[(size_t)n * C3 + d]));
  }
  red[threadIdx.x] = m;
  __syncthreads();
  for (int s2 = 128; s2 > 0; s2 >>= 1) {
    if (threadIdx.x < s2) red[threadIdx.x] = fmaxf(red[threadIdx.x], red[threadIdx.x + s2]);
    __syncthreads();
  }
  if (threadIdx.x == 0) scales[which] = 448.0f / fmaxf(red[0], 1e-12f);
}

__global__ __launch_bounds__(256) void quant_kernel(const float* __restrict__ qkv,
                                                    const float* __restrict__ scales,
                                                    uint8_t* __restrict__ q8,
                                                    uint8_t* __restrict__ k8,
                                                    _Float16* __restrict__ vt) {
  size_t i = (size_t)blockIdx.x * 256 + threadIdx.x;
  if (i >= (size_t)BB * HH * NN * DD) return;
  int d = (int)(i & 63);
  size_t r = i >> 6;
  int n = (int)(r % NN);
  size_t bh = r / NN;
  int b = (int)(bh / HH), hh = (int)(bh - (size_t)b * HH);
  size_t src = ((size_t)(b * NN + n)) * C3 + hh * DD + d;
  float sq = scales[bh * 2 + 0], sk = scales[bh * 2 + 1];
  q8[i] = f32_to_e4m3(qkv[src] * sq);
  k8[i] = f32_to_e4m3(qkv[src + CC] * sk);
  vt[(bh * DD + d) * NN + n] = (_Float16)qkv[src + 2 * CC];
}

__global__ __launch_bounds__(32) void flash_attn_fp8(const uint8_t* __restrict__ q8,
                                                     const uint8_t* __restrict__ k8,
                                                     const _Float16* __restrict__ vt,
                                                     const float* __restrict__ scales,
                                                     __bf16* __restrict__ out) {
  __shared__ alignas(16) _Float16 pbuf[32 * 32];
  int lane = threadIdx.x;
  int l15 = lane & 15, h = lane >> 4;
  int mt = blockIdx.x & (NN / 32 - 1);
  int bh = blockIdx.x / (NN / 32);
  int b = bh / HH, hh = bh - b * HH;
  int m0 = mt * 32;
  float sq = scales[bh * 2 + 0], sk = scales[bh * 2 + 1];
  float qk_scale = 0.125f / (sq * sk);

  v8i qa[2];
#pragma unroll
  for (int g = 0; g < 2; ++g) {
    const uint8_t* qb = q8 + ((size_t)bh * NN + m0 + g * 16 + l15) * DD;
    uint2 t0 = *(const uint2*)(qb + h * 8);
    uint2 t1 = *(const uint2*)(qb + 16 + h * 8);
    uint2 t2 = *(const uint2*)(qb + 32 + h * 8);
    uint2 t3 = *(const uint2*)(qb + 48 + h * 8);
    qa[g][0] = t0.x; qa[g][1] = t0.y; qa[g][2] = t1.x; qa[g][3] = t1.y;
    qa[g][4] = t2.x; qa[g][5] = t2.y; qa[g][6] = t3.x; qa[g][7] = t3.y;
  }

  float mrow[2][8], lrow[2][8];
  v8f O[2][4];
  v8f zero = {};
#pragma unroll
  for (int g = 0; g < 2; ++g) {
#pragma unroll
    for (int t = 0; t < 4; ++t) O[g][t] = zero;
#pragma unroll
    for (int r = 0; r < 8; ++r) { mrow[g][r] = -1e30f; lrow[g][r] = 0.0f; }
  }

  const int kend = m0 + 32;
  for (int kk = 0; kk < kend; kk += 32) {
    if (kk + 32 < kend) {
      __builtin_prefetch(k8 + ((size_t)bh * NN + kk + 32 + l15) * DD, 0, 3);
      __builtin_prefetch(vt + ((size_t)bh * DD + l15 * 4) * NN + kk + 32, 0, 3);
    }

    v8i kb[2];
#pragma unroll
    for (int t = 0; t < 2; ++t) {
      const uint8_t* kr = k8 + ((size_t)bh * NN + kk + t * 16 + l15) * DD;
      uint2 u0 = *(const uint2*)(kr + h * 8);
      uint2 u1 = *(const uint2*)(kr + 16 + h * 8);
      uint2 u2 = *(const uint2*)(kr + 32 + h * 8);
      uint2 u3 = *(const uint2*)(kr + 48 + h * 8);
      kb[t][0] = u0.x; kb[t][1] = u0.y; kb[t][2] = u1.x; kb[t][3] = u1.y;
      kb[t][4] = u2.x; kb[t][5] = u2.y; kb[t][6] = u3.x; kb[t][7] = u3.y;
    }

    v8f S[2][2];
#pragma unroll
    for (int g = 0; g < 2; ++g)
#pragma unroll
      for (int t = 0; t < 2; ++t)
        S[g][t] = __builtin_amdgcn_wmma_f32_16x16x64_fp8_fp8(qa[g], kb[t], (short)0,
                                                             zero, false, false);

#pragma unroll
    for (int g = 0; g < 2; ++g) {
#pragma unroll
      for (int r = 0; r < 8; ++r) {
        int rowg = m0 + g * 16 + r + 8 * h;
        float s0 = S[g][0][r] * qk_scale;
        if (kk + l15 > rowg) s0 = -1e30f;
        float s1 = S[g][1][r] * qk_scale;
        if (kk + 16 + l15 > rowg) s1 = -1e30f;
        float mx = fmaxf(s0, s1);
        mx = fmaxf(mx, __shfl_xor(mx, 1, 32));
        mx = fmaxf(mx, __shfl_xor(mx, 2, 32));
        mx = fmaxf(mx, __shfl_xor(mx, 4, 32));
        mx = fmaxf(mx, __shfl_xor(mx, 8, 32));
        float mnew  = fmaxf(mrow[g][r], mx);
        float alpha = __expf(mrow[g][r] - mnew);
        mrow[g][r] = mnew;
        float p0 = __expf(s0 - mnew);
        float p1 = __expf(s1 - mnew);
        float rs = p0 + p1;
        rs += __shfl_xor(rs, 1, 32);
        rs += __shfl_xor(rs, 2, 32);
        rs += __shfl_xor(rs, 4, 32);
        rs += __shfl_xor(rs, 8, 32);
        lrow[g][r] = lrow[g][r] * alpha + rs;
        O[g][0][r] *= alpha; O[g][1][r] *= alpha;
        O[g][2][r] *= alpha; O[g][3][r] *= alpha;
        int row = g * 16 + r + 8 * h;
        pbuf[row * 32 + l15]      = (_Float16)p0;
        pbuf[row * 32 + 16 + l15] = (_Float16)p1;
      }
    }
    __syncthreads();

    U16h pa[2];
#pragma unroll
    for (int g = 0; g < 2; ++g) {
      const _Float16* pr = pbuf + (g * 16 + l15) * 32;
      pa[g].u[0] = *(const uint4*)(pr + h * 8);
      pa[g].u[1] = *(const uint4*)(pr + 16 + h * 8);
    }

    const _Float16* vb0 = vt + ((size_t)bh * DD) * NN + kk + h * 8;
#pragma unroll
    for (int t = 0; t < 4; ++t) {
      U16h vb;
      const _Float16* vr = vb0 + (size_t)(t * 16 + l15) * NN;
      vb.u[0] = *(const uint4*)vr;
      vb.u[1] = *(const uint4*)(vr + 16);
      O[0][t] = __builtin_amdgcn_wmma_f32_16x16x32_f16(false, pa[0].v, false, vb.v,
                                                       (short)0, O[0][t], false, false);
      O[1][t] = __builtin_amdgcn_wmma_f32_16x16x32_f16(false, pa[1].v, false, vb.v,
                                                       (short)0, O[1][t], false, false);
    }
    __syncthreads();
  }

#pragma unroll
  for (int g = 0; g < 2; ++g) {
#pragma unroll
    for (int r = 0; r < 8; ++r) {
      float inv = 1.0f / lrow[g][r];
      int row = m0 + g * 16 + r + 8 * h;
      size_t base = (size_t)(b * NN + row) * CC + hh * DD + l15;
      out[base + 0]  = (__bf16)(O[g][0][r] * inv);
      out[base + 16] = (__bf16)(O[g][1][r] * inv);
      out[base + 32] = (__bf16)(O[g][2][r] * inv);
      out[base + 48] = (__bf16)(O[g][3][r] * inv);
    }
  }
}

extern "C" void kernel_launch(void* const* d_in, const int* in_sizes, int n_in,
                              void* d_out, int out_size, void* d_ws, size_t ws_size,
                              hipStream_t stream) {
  const float* x      = (const float*)d_in[0];
  const float* w_qkv  = (const float*)d_in[1];
  const float* w_proj = (const float*)d_in[2];
  const float* b_proj = (const float*)d_in[3];
  float* out = (float*)d_out;

  char* ws = (char*)d_ws;
  size_t off = 0;
  auto take = [&](size_t bytes) -> char* {
    char* p = ws + off;
    off = (off + bytes + 255) & ~(size_t)255;
    return p;
  };
  float*    qkv    = (float*)take((size_t)BB * NN * C3 * 4);
  __bf16*   xbf    = (__bf16*)take((size_t)BB * NN * CC * 2);
  __bf16*   wqkvT  = (__bf16*)take((size_t)C3 * CC * 2);
  __bf16*   wprojT = (__bf16*)take((size_t)CC * CC * 2);
  float*    scales = (float*)take((size_t)BB * HH * 2 * 4);
  uint8_t*  q8     = (uint8_t*)take((size_t)BB * HH * NN * DD);
  uint8_t*  k8     = (uint8_t*)take((size_t)BB * HH * NN * DD);
  _Float16* vtb    = (_Float16*)take((size_t)BB * HH * NN * DD * 2);
  __bf16*   attn   = (__bf16*)take((size_t)BB * NN * CC * 2);

  {
    int n = BB * NN * CC;
    cvt_f32_bf16<<<n / 256, 256, 0, stream>>>(x, xbf, n);
  }
  cvt_transpose_bf16<<<(CC * C3) / 256, 256, 0, stream>>>(w_qkv, wqkvT, CC, C3);
  cvt_transpose_bf16<<<(CC * CC) / 256, 256, 0, stream>>>(w_proj, wprojT, CC, CC);

  {
    int M = BB * NN, Nc = C3, K = CC;
    int tiles = (M / 64) * (Nc / 64);
    gemm_bf16_wmma<<<(tiles + 7) / 8, 256, 0, stream>>>(xbf, wqkvT, qkv, nullptr, M, Nc, K);
  }

  amax_scale_kernel<<<BB * HH * 2, 256, 0, stream>>>(qkv, scales);
  {
    size_t total = (size_t)BB * HH * NN * DD;
    quant_kernel<<<(unsigned)((total + 255) / 256), 256, 0, stream>>>(qkv, scales, q8, k8, vtb);
  }

  flash_attn_fp8<<<BB * HH * (NN / 32), 32, 0, stream>>>(q8, k8, vtb, scales, attn);

  {
    int M = BB * NN, Nc = CC, K = CC;
    int tiles = (M / 64) * (Nc / 64);
    gemm_bf16_wmma<<<(tiles + 7) / 8, 256, 0, stream>>>(attn, wprojT, out, b_proj, M, Nc, K);
  }
  (void)in_sizes; (void)n_in; (void)out_size; (void)ws_size;
}
